// DGCN_9328668967434
// MI455X (gfx1250) — matrix operand order repaired and checked by reading, not compiled, not run on hardware
//
#include <hip/hip_runtime.h>
#include <hip/hip_bf16.h>

// Dims from reference: B=64, N=512, W=24, F=16, H=128, K=2
#define BB 64
#define NN 512
#define WW 24
#define FF 16
#define HH 128

typedef __attribute__((ext_vector_type(16))) __bf16 v16bf;
typedef __attribute__((ext_vector_type(8)))  __bf16 v8bf;
typedef __attribute__((ext_vector_type(8)))  float  v8f;

#define CAT16(lo,hi) __builtin_shufflevector((lo),(hi),0,1,2,3,4,5,6,7,8,9,10,11,12,13,14,15)
#define WMMA_BF16(Am,Bm,Cm) __builtin_amdgcn_wmma_f32_16x16x32_bf16(false,(Am),false,(Bm),(short)0,(Cm),false,false)

__device__ __forceinline__ float sigmoidf_(float v) { return 1.0f / (1.0f + __expf(-v)); }

// ---------------------------------------------------------------------------
// Prep 1: A -> bf16 row-major; merged dc weights -> WcatT[h][64] (B-operand
// col-major, K padded 48->64, hop weights merged since fw==bw); WoutT[f][128].
// ---------------------------------------------------------------------------
__global__ __launch_bounds__(256) void prep_misc_kernel(
    const float* __restrict__ A, const float* __restrict__ dcw,
    const float* __restrict__ Wout,
    __bf16* __restrict__ A_bf, __bf16* __restrict__ WcatT,
    __bf16* __restrict__ WoutT) {
  int i = blockIdx.x * 256 + threadIdx.x;
  if (i < NN * NN) {
    A_bf[i] = (__bf16)A[i];
    return;
  }
  int j = i - NN * NN;
  if (j < HH * 64) {
    int h = j >> 6, k = j & 63;
    float v = 0.0f;
    if (k < 16)      v = dcw[k * HH + h];                                 // x0
    else if (k < 32) v = dcw[k * HH + h] + dcw[(k + 16) * HH + h];        // x1 fw+bw
    else if (k < 48) v = dcw[(k + 16) * HH + h] + dcw[(k + 32) * HH + h]; // x2 fw+bw
    WcatT[h * 64 + k] = (__bf16)v;
    return;
  }
  int m = j - HH * 64;
  if (m < FF * HH) {
    int f = m >> 7, h = m & 127;
    WoutT[f * HH + h] = (__bf16)Wout[h * FF + f];  // W_out^T in B-operand layout
  }
}

// ---------------------------------------------------------------------------
// Prep 2: W_ih/W_hh fp32 -> bf16 (layout [N][3H][H] kept: it is already the
// B-operand layout for gi/gh GEMMs since (W^T) col-major == W row-major).
// ---------------------------------------------------------------------------
__global__ __launch_bounds__(256) void prep_w_kernel(
    const float* __restrict__ Wih, const float* __restrict__ Whh,
    __bf16* __restrict__ Wih_bf, __bf16* __restrict__ Whh_bf) {
  const int total = NN * 3 * HH * HH;  // 25,165,824
  for (int i = blockIdx.x * 256 + threadIdx.x; i < total; i += gridDim.x * 256) {
    Wih_bf[i] = (__bf16)Wih[i];
    Whh_bf[i] = (__bf16)Whh[i];
  }
}

// ---------------------------------------------------------------------------
// Diffusion conv + dc projection. One workgroup per (b,w) slice (1536 WGs).
//   X1 = A @ X0 ; X2 = A @ X1 ; SP = [X0|X1|X2|0] @ Wcat   (all bf16 WMMA)
// LDS (dynamic, 96KB): X0T[16][512], X1T[16][512] (B-operand col-major),
// Xcat[512][64] (A-operand row-major). A_bf (512KB) streams from L2 and is
// reused by all 1536 workgroups. Writes SPN[n][t*64+b][h] bf16.
// ---------------------------------------------------------------------------
__global__ __launch_bounds__(256) void diffusion_kernel(
    const float* __restrict__ x, const __bf16* __restrict__ A_bf,
    const __bf16* __restrict__ WcatT, __bf16* __restrict__ SPN) {
  extern __shared__ __bf16 smem[];
  __bf16* X0T  = smem;                 // 16*512
  __bf16* X1T  = smem + FF * NN;       // 16*512
  __bf16* Xcat = smem + 2 * FF * NN;   // 512*64

  const int tid = threadIdx.x;
  const int bw = blockIdx.x;
  const int b = bw / WW, w = bw % WW;

  // Stage X0 (gather from x[B][N][W][F]) into both layouts; zero pad cols.
  for (int idx = tid; idx < NN * FF; idx += 256) {
    int n = idx >> 4, f = idx & 15;
    float v = x[(((size_t)b * NN + n) * WW + w) * FF + f];
    __bf16 bv = (__bf16)v;
    X0T[f * NN + n] = bv;
    Xcat[n * 64 + f] = bv;
    Xcat[n * 64 + 48 + f] = (__bf16)0.0f;
  }
  __syncthreads();

  const int lane = tid & 31, wv = tid >> 5;
  const int lr = lane & 15, hs = lane >> 4;

  // ---- hop 1: X1 = A @ X0 --------------------------------------------------
  for (int mi = 0; mi < 4; ++mi) {
    int mt = wv * 4 + mi;
    v8f c = {};
    const __bf16* Arow = A_bf + (size_t)(mt * 16 + lr) * NN;
    for (int kk = 0; kk < 16; ++kk) {
      int k0 = kk * 32 + hs * 16;
      v8bf alo = *(const v8bf*)(Arow + k0);
      v8bf ahi = *(const v8bf*)(Arow + k0 + 8);
      v16bf af = CAT16(alo, ahi);
      v16bf bfr = *(const v16bf*)(X0T + lr * NN + kk * 32 + hs * 16);
      c = WMMA_BF16(af, bfr, c);
    }
    int nb = mt * 16 + hs * 8;
    v8bf pk;
#pragma unroll
    for (int g = 0; g < 8; ++g) pk[g] = (__bf16)c[g];
    *(v8bf*)(X1T + lr * NN + nb) = pk;   // col-major for next hop's B operand
#pragma unroll
    for (int g = 0; g < 8; ++g) Xcat[(nb + g) * 64 + 16 + lr] = pk[g];
  }
  __syncthreads();

  // ---- hop 2: X2 = A @ X1 --------------------------------------------------
  for (int mi = 0; mi < 4; ++mi) {
    int mt = wv * 4 + mi;
    v8f c = {};
    const __bf16* Arow = A_bf + (size_t)(mt * 16 + lr) * NN;
    for (int kk = 0; kk < 16; ++kk) {
      int k0 = kk * 32 + hs * 16;
      v8bf alo = *(const v8bf*)(Arow + k0);
      v8bf ahi = *(const v8bf*)(Arow + k0 + 8);
      v16bf af = CAT16(alo, ahi);
      v16bf bfr = *(const v16bf*)(X1T + lr * NN + kk * 32 + hs * 16);
      c = WMMA_BF16(af, bfr, c);
    }
    int nb = mt * 16 + hs * 8;
#pragma unroll
    for (int g = 0; g < 8; ++g) Xcat[(nb + g) * 64 + 32 + lr] = (__bf16)c[g];
  }
  __syncthreads();

  // ---- SP = Xcat[512x64] @ Wcat[64x128] ------------------------------------
  const int row = w * BB + b;  // t*B + b ordering inside SPN
  for (int ti = wv; ti < 256; ti += 8) {
    int mt = ti >> 3, nt = ti & 7;
    v8f c = {};
    const __bf16* xrow = Xcat + (mt * 16 + lr) * 64;
    const __bf16* wcol = WcatT + (nt * 16 + lr) * 64;
#pragma unroll
    for (int kk = 0; kk < 2; ++kk) {
      int k0 = kk * 32 + hs * 16;
      v8bf alo = *(const v8bf*)(xrow + k0);
      v8bf ahi = *(const v8bf*)(xrow + k0 + 8);
      v16bf af = CAT16(alo, ahi);
      v16bf bfr = *(const v16bf*)(wcol + kk * 32 + hs * 16);
      c = WMMA_BF16(af, bfr, c);
    }
    int h = nt * 16 + lr;
#pragma unroll
    for (int g = 0; g < 8; ++g) {
      int node = mt * 16 + hs * 8 + g;
      SPN[((size_t)node * (WW * BB) + row) * HH + h] = (__bf16)c[g];
    }
  }
}

// ---------------------------------------------------------------------------
// Per-node GRU recurrence + fused output projection. One workgroup per node.
// Wave wv owns h-columns [wv*16, wv*16+16); all 24 weight B-fragments for both
// W_ih and W_hh are loop-invariant and preloaded into VGPRs (wave32 budget).
// h state (64x128 bf16) lives in LDS and is re-read as WMMA A fragments.
// The 24-step recurrence is the only serial region: prefetch t+1's SPN block
// (global_prefetch) while computing step t to hide HBM/L2 latency.
// ---------------------------------------------------------------------------
__global__ __launch_bounds__(256) void gru_kernel(
    const __bf16* __restrict__ SPN, const __bf16* __restrict__ Wih_bf,
    const __bf16* __restrict__ Whh_bf, const float* __restrict__ b_ih,
    const float* __restrict__ b_hh, const __bf16* __restrict__ WoutT,
    const float* __restrict__ b_out, float* __restrict__ out) {
  __shared__ __bf16 Hst[BB * HH];  // 16 KB

  const int tid = threadIdx.x, lane = tid & 31, wv = tid >> 5;
  const int lr = lane & 15, hs = lane >> 4;
  const int n = blockIdx.x;
  const int h0 = wv * 16;

  for (int i = tid; i < BB * HH; i += 256) Hst[i] = (__bf16)0.0f;

  // Preload invariant weight fragments: gates {r,z,n} x K-steps {0..3}.
  v16bf wif[3][4], whf[3][4];
  const size_t wbase = (size_t)n * 3 * HH * HH;
#pragma unroll
  for (int g3 = 0; g3 < 3; ++g3) {
#pragma unroll
    for (int kk = 0; kk < 4; ++kk) {
      size_t off = wbase + (size_t)(g3 * HH + h0 + lr) * HH + kk * 32 + hs * 16;
      wif[g3][kk] = *(const v16bf*)(Wih_bf + off);
      whf[g3][kk] = *(const v16bf*)(Whh_bf + off);
    }
  }
  float bi[3], bh[3];
#pragma unroll
  for (int g3 = 0; g3 < 3; ++g3) {
    bi[g3] = b_ih[n * 3 * HH + g3 * HH + h0 + lr];
    bh[g3] = b_hh[n * 3 * HH + g3 * HH + h0 + lr];
  }
  __syncthreads();

  const __bf16* spn = SPN + (size_t)n * (WW * BB) * HH;
  float hnew[4][8];

  // Prefetch timestep 0's input block (16 KB = 256 threads x 1 line).
  __builtin_prefetch(spn + (size_t)tid * 64, 0, 1);

  for (int t = 0; t < WW; ++t) {
    // Prefetch next timestep's 64x128 bf16 block while computing this one.
    if (t + 1 < WW)
      __builtin_prefetch(spn + (size_t)(t + 1) * BB * HH + (size_t)tid * 64, 0, 1);
#pragma unroll
    for (int mb = 0; mb < 4; ++mb) {
      v8f cr = {}, cz = {}, cn = {}, dr = {}, dz = {}, dn = {};
#pragma unroll
      for (int kk = 0; kk < 4; ++kk) {
        const __bf16* xr = spn + (size_t)(t * BB + mb * 16 + lr) * HH + kk * 32 + hs * 16;
        v8bf xlo = *(const v8bf*)xr;
        v8bf xhi = *(const v8bf*)(xr + 8);
        v16bf ax = CAT16(xlo, xhi);
        const __bf16* hr = Hst + (mb * 16 + lr) * HH + kk * 32 + hs * 16;
        v8bf hlo = *(const v8bf*)hr;
        v8bf hhi = *(const v8bf*)(hr + 8);
        v16bf ah = CAT16(hlo, hhi);
        cr = WMMA_BF16(ax, wif[0][kk], cr);
        cz = WMMA_BF16(ax, wif[1][kk], cz);
        cn = WMMA_BF16(ax, wif[2][kk], cn);
        dr = WMMA_BF16(ah, whf[0][kk], dr);
        dz = WMMA_BF16(ah, whf[1][kk], dz);
        dn = WMMA_BF16(ah, whf[2][kk], dn);
      }
#pragma unroll
      for (int g = 0; g < 8; ++g) {
        int brow = mb * 16 + hs * 8 + g;
        float hprev = (float)Hst[brow * HH + h0 + lr];
        float r = sigmoidf_(cr[g] + bi[0] + dr[g] + bh[0]);
        float z = sigmoidf_(cz[g] + bi[1] + dz[g] + bh[1]);
        float nn_ = tanhf(cn[g] + bi[2] + r * (dn[g] + bh[2]));
        hnew[mb][g] = (1.0f - z) * nn_ + z * hprev;
      }
    }
    __syncthreads();  // all waves done reading Hst
#pragma unroll
    for (int mb = 0; mb < 4; ++mb) {
#pragma unroll
      for (int g = 0; g < 8; ++g) {
        int brow = mb * 16 + hs * 8 + g;
        Hst[brow * HH + h0 + lr] = (__bf16)hnew[mb][g];
      }
    }
    __syncthreads();  // new h visible
  }

  // Fused output projection: out = hT @ W_out + b_out  (M=64, K=128, N=16)
  if (wv < 4) {
    int mb = wv;
    v8f co = {};
#pragma unroll
    for (int kk = 0; kk < 4; ++kk) {
      const __bf16* hr = Hst + (mb * 16 + lr) * HH + kk * 32 + hs * 16;
      v8bf hlo = *(const v8bf*)hr;
      v8bf hhi = *(const v8bf*)(hr + 8);
      v16bf ah = CAT16(hlo, hhi);
      v16bf bo = *(const v16bf*)(WoutT + lr * HH + kk * 32 + hs * 16);
      co = WMMA_BF16(ah, bo, co);
    }
    float bv = b_out[lr];
#pragma unroll
    for (int g = 0; g < 8; ++g) {
      int b = mb * 16 + hs * 8 + g;
      out[((size_t)b * NN + n) * FF + lr] = co[g] + bv;
    }
  }
}

// ---------------------------------------------------------------------------
extern "C" void kernel_launch(void* const* d_in, const int* in_sizes, int n_in,
                              void* d_out, int out_size, void* d_ws, size_t ws_size,
                              hipStream_t stream) {
  const float* x    = (const float*)d_in[0];
  const float* A    = (const float*)d_in[1];
  const float* dcw  = (const float*)d_in[2];
  const float* Wih  = (const float*)d_in[3];
  const float* Whh  = (const float*)d_in[4];
  const float* bih  = (const float*)d_in[5];
  const float* bhh  = (const float*)d_in[6];
  const float* Wout = (const float*)d_in[7];
  const float* bout = (const float*)d_in[8];
  float* out = (float*)d_out;

  char* ws = (char*)d_ws;
  // Workspace layout (bytes):
  //   A_bf   @ 0x0        512 KB
  //   WcatT  @ 0x80000     16 KB
  //   WoutT  @ 0x84000      4 KB
  //   Wih_bf @ 0x100000    48 MB
  //   Whh_bf @ 0x3100000   48 MB
  //   SPN    @ 0x6100000  192 MB   (total ~303 MB)
  __bf16* A_bf   = (__bf16*)(ws + 0x0);
  __bf16* WcatT  = (__bf16*)(ws + 0x80000);
  __bf16* WoutT  = (__bf16*)(ws + 0x84000);
  __bf16* Wih_bf = (__bf16*)(ws + 0x100000);
  __bf16* Whh_bf = (__bf16*)(ws + 0x100000 + 0x3000000);
  __bf16* SPN    = (__bf16*)(ws + 0x100000 + 0x6000000);

  prep_misc_kernel<<<1064, 256, 0, stream>>>(A, dcw, Wout, A_bf, WcatT, WoutT);
  prep_w_kernel<<<4096, 256, 0, stream>>>(Wih, Whh, Wih_bf, Whh_bf);

  const size_t dshm = (size_t)(2 * FF * NN + NN * 64) * sizeof(__bf16);  // 96 KB
  hipFuncSetAttribute((const void*)diffusion_kernel,
                      hipFuncAttributeMaxDynamicSharedMemorySize, (int)dshm);
  diffusion_kernel<<<BB * WW, 256, dshm, stream>>>(x, A_bf, WcatT, SPN);

  gru_kernel<<<NN, 256, 0, stream>>>(SPN, Wih_bf, Whh_bf, bih, bhh, WoutT, bout, out);
}
